// Multi_Head_SelfAttention_7851200217461
// MI455X (gfx1250) — compile-verified, matrix operand order checked by reading, not run on hardware
//
#include <hip/hip_runtime.h>
#include <hip/hip_bf16.h>

// ---------------------------------------------------------------------------
// Multi-head self-attention (B=1, S=2048, E=2048, H=16, D=128) for gfx1250.
// All GEMMs routed through v_wmma_f32_16x16x32_bf16 (f32 accumulate).
// Projection GEMMs stage tiles global -> LDS with GLOBAL_LOAD_ASYNC_TO_LDS
// (ASYNCcnt path) when the builtin is available; ds_load_b128 fragments.
// Workspace layout (needs 80 MB of d_ws):
//   [0..8M)   Xb   bf16 [S,E]
//   [8..40M)  Wqb/Wkb/Wvb/Wob bf16 [E,E] each 8MB
//   [40..64M) Qb/Kb/Vb bf16 [S,E]
//   [64..72M) Vt   bf16 [H][128][S]
//   [72..80M) Ctx  bf16 [S,E]
// ---------------------------------------------------------------------------

#define SEQ 2048
#define EMB 2048
#define NH  16
#define HD  128

typedef unsigned short u16;
typedef unsigned int   u32;
typedef __bf16 v16bf __attribute__((ext_vector_type(16)));
typedef float  v8f   __attribute__((ext_vector_type(8)));
typedef float  v4f   __attribute__((ext_vector_type(4)));
typedef u32    v4u   __attribute__((ext_vector_type(4)));
typedef u32    v2u   __attribute__((ext_vector_type(2)));

#if defined(__has_builtin)
#if __has_builtin(__builtin_amdgcn_global_load_async_to_lds_b128) && \
    __has_builtin(__builtin_amdgcn_s_wait_asynccnt)
#define USE_ASYNC_COPY 1
#endif
#endif
#ifndef USE_ASYNC_COPY
#define USE_ASYNC_COPY 0
#endif

#if USE_ASYNC_COPY
// Builtin prototype (from hipcc diagnostic): parameters are pointers to a
// 4 x i32 GCC vector, global (AS1) source and LDS (AS3) destination.
typedef int v4i_gcc __attribute__((vector_size(16)));
typedef __attribute__((address_space(1))) v4i_gcc* as1_v4i_ptr;
typedef __attribute__((address_space(3))) v4i_gcc* as3_v4i_ptr;
#endif

__device__ __forceinline__ u16 f2bf(float f) {
  union { float f; u32 u; } c; c.f = f;
  u32 u = c.u + 0x7FFFu + ((c.u >> 16) & 1u);   // round-to-nearest-even
  return (u16)(u >> 16);
}
__device__ __forceinline__ float bf2f(u16 h) {
  union { u32 u; float f; } c; c.u = ((u32)h) << 16;
  return c.f;
}

// Load one 16-element bf16 fragment per lane following the CDNA5 16-bit
// A/B layout: elements 0..7 at p, elements 8..15 at p+16 elements.
// (p must already include the +8*laneHalf K offset; 16B aligned.)
__device__ __forceinline__ v16bf load_frag(const u16* p) {
  union { v16bf v; v4u u[2]; } t;
  t.u[0] = *(const v4u*)(p);
  t.u[1] = *(const v4u*)(p + 16);
  return t.v;
}

__device__ __forceinline__ v8f zero_v8f() {
  v8f z;
#pragma unroll
  for (int i = 0; i < 8; ++i) z[i] = 0.0f;
  return z;
}

__device__ __forceinline__ v8f wmma_bf16(v16bf a, v16bf b, v8f c) {
  return __builtin_amdgcn_wmma_f32_16x16x32_bf16(
      false, a, false, b, (short)0, c, false, false);
}

// One 16B global->LDS copy: async (ASYNCcnt) if available, else sync.
__device__ __forceinline__ void stage_b128(const u16* g, u16* l) {
#if USE_ASYNC_COPY
  __builtin_amdgcn_global_load_async_to_lds_b128(
      (as1_v4i_ptr)g, (as3_v4i_ptr)l, 0, 0);
#else
  *(v4u*)l = *(const v4u*)g;
#endif
}
__device__ __forceinline__ void wait_stage() {
#if USE_ASYNC_COPY
  __builtin_amdgcn_s_wait_asynccnt(0);
#endif
}

// ---------------------------------------------------------------------------
// 1) f32 -> bf16 conversion (vectorized x4)
// ---------------------------------------------------------------------------
__global__ __launch_bounds__(256) void cvt_bf16(const float* __restrict__ in,
                                                u16* __restrict__ out, int n) {
  int i = (blockIdx.x * 256 + threadIdx.x) * 4;
  if (i >= n) return;
  v4f f = *(const v4f*)(in + i);
  union { v2u u; u16 s[4]; } o;
  o.s[0] = f2bf(f[0]); o.s[1] = f2bf(f[1]);
  o.s[2] = f2bf(f[2]); o.s[3] = f2bf(f[3]);
  *(v2u*)(out + i) = o.u;
}

// ---------------------------------------------------------------------------
// 2) GEMM-NT: C[M,N] = A[M,K] * B[N,K]^T   (A, B bf16 row-major, K-contig)
//    Block = 256 threads = 8 waves; block tile 128x128, wave tile 64x32.
//    Double-buffered LDS staging (K-chunk 32) via async global->LDS copies.
//    LDS rows padded to 40 bf16 (80B): 16 fragment rows -> 16 distinct
//    bank groups, conflict-free ds_load_b128.
// ---------------------------------------------------------------------------
template <bool OUT_BF16>
__global__ __launch_bounds__(256) void gemm_nt(const u16* __restrict__ A,
                                               const u16* __restrict__ B,
                                               void* __restrict__ Cv,
                                               int M, int N, int K) {
  __shared__ __align__(16) u16 As[2][128][40];
  __shared__ __align__(16) u16 Bs[2][128][40];

  const int tid  = threadIdx.x;
  const int lane = tid & 31;
  const int wid  = tid >> 5;
  const int lm = lane & 15, lh = lane >> 4;
  const int m0b = blockIdx.y * 128;
  const int n0b = blockIdx.x * 128;
  const int mw = (wid & 1) * 64;   // wave M offset in block tile
  const int nw = (wid >> 1) * 32;  // wave N offset in block tile

  // Staging map: 128 rows x 4 b128-chunks per tile = 512 chunks;
  // each of 256 threads stages 2 chunks of A and 2 of B per K-chunk.
  const int sr = tid >> 2;         // row base 0..63 (step 64)
  const int sk = (tid & 3) * 8;    // k offset 0/8/16/24

  auto stage = [&](int buf, int k0) {
#pragma unroll
    for (int c = 0; c < 2; ++c) {
      int r = sr + c * 64;
      stage_b128(A + (size_t)(m0b + r) * K + k0 + sk, &As[buf][r][sk]);
      stage_b128(B + (size_t)(n0b + r) * K + k0 + sk, &Bs[buf][r][sk]);
    }
  };

  v8f acc[4][2];
#pragma unroll
  for (int i = 0; i < 4; ++i)
#pragma unroll
    for (int j = 0; j < 2; ++j) acc[i][j] = zero_v8f();

  stage(0, 0);
  int cur = 0;
  for (int k0 = 0; k0 < K; k0 += 32) {
    wait_stage();          // own async copies for buffer `cur` complete
    __syncthreads();       // everyone's copies complete, prev reads done
    if (k0 + 32 < K) stage(cur ^ 1, k0 + 32);

    v16bf a[4], b[2];
#pragma unroll
    for (int i = 0; i < 4; ++i)
      a[i] = load_frag(&As[cur][mw + i * 16 + lm][lh * 8]);
#pragma unroll
    for (int j = 0; j < 2; ++j)
      b[j] = load_frag(&Bs[cur][nw + j * 16 + lm][lh * 8]);
#pragma unroll
    for (int i = 0; i < 4; ++i)
#pragma unroll
      for (int j = 0; j < 2; ++j)
        acc[i][j] = wmma_bf16(a[i], b[j], acc[i][j]);
    cur ^= 1;
  }

  // D layout: lane holds column n=lm; VGPR r holds row m = r + 8*lh.
  const int m0 = m0b + mw, n0 = n0b + nw;
#pragma unroll
  for (int i = 0; i < 4; ++i)
#pragma unroll
    for (int j = 0; j < 2; ++j)
#pragma unroll
      for (int r = 0; r < 8; ++r) {
        int m = m0 + i * 16 + r + 8 * lh;
        int n = n0 + j * 16 + lm;
        if (OUT_BF16) ((u16*)Cv)[(size_t)m * N + n] = f2bf(acc[i][j][r]);
        else          ((float*)Cv)[(size_t)m * N + n] = acc[i][j][r];
      }
}

// ---------------------------------------------------------------------------
// 3) RoPE applied in-place to bf16 Q and K ([S, H*D] layout)
// ---------------------------------------------------------------------------
__global__ __launch_bounds__(256) void rope_kernel(u16* __restrict__ Q,
                                                   u16* __restrict__ Kb) {
  int tid = blockIdx.x * 256 + threadIdx.x;
  if (tid >= SEQ * NH * (HD / 2)) return;
  int d = tid & 63;
  int h = (tid >> 6) & (NH - 1);
  int s = tid >> 10;
  float inv_freq = __powf(10000.0f, -(float)d / 64.0f);
  float ang = (float)s * inv_freq;
  float c = cosf(ang), sn = sinf(ang);
  size_t base = (size_t)s * EMB + h * HD + d;
  float q0 = bf2f(Q[base]), q1 = bf2f(Q[base + 64]);
  Q[base]      = f2bf(q0 * c - q1 * sn);
  Q[base + 64] = f2bf(q1 * c + q0 * sn);
  float k0 = bf2f(Kb[base]), k1 = bf2f(Kb[base + 64]);
  Kb[base]      = f2bf(k0 * c - k1 * sn);
  Kb[base + 64] = f2bf(k1 * c + k0 * sn);
}

// ---------------------------------------------------------------------------
// 4) Per-head transpose V[S, H*D] -> Vt[H][D][S]  (LDS-tiled, 32x32)
// ---------------------------------------------------------------------------
__global__ __launch_bounds__(256) void transpose_v(const u16* __restrict__ V,
                                                   u16* __restrict__ Vt) {
  __shared__ u16 tile[32][33];
  int h  = blockIdx.y >> 2;
  int dt = blockIdx.y & 3;
  int s0 = blockIdx.x * 32;
  int tx = threadIdx.x, ty = threadIdx.y;  // 32 x 8
#pragma unroll
  for (int j = 0; j < 4; ++j) {
    int s = s0 + ty + j * 8;
    tile[ty + j * 8][tx] = V[(size_t)s * EMB + h * HD + dt * 32 + tx];
  }
  __syncthreads();
#pragma unroll
  for (int j = 0; j < 4; ++j) {
    int d = dt * 32 + ty + j * 8;
    Vt[((size_t)h * HD + d) * SEQ + s0 + tx] = tile[tx][ty + j * 8];
  }
}

// ---------------------------------------------------------------------------
// 5) Flash attention (online softmax). One wave per (head, 16-query tile).
//    S^T = K_tile x Q^T so score D-layout == P B-fragment layout (no
//    cross-lane shuffles between the two WMMAs).
// ---------------------------------------------------------------------------
__global__ __launch_bounds__(128) void attn_kernel(const u16* __restrict__ Q,
                                                   const u16* __restrict__ Kc,
                                                   const u16* __restrict__ Vt,
                                                   u16* __restrict__ Ctx) {
  const int lane = threadIdx.x & 31;
  const int wid  = threadIdx.x >> 5;
  const int lm = lane & 15, lh = lane >> 4;
  const int h  = blockIdx.y;
  const int q0 = (blockIdx.x * 4 + wid) * 16;

  // Q as B-operand: lane = query column lm, K-dim = head dim.
  v16bf qf[4];
  {
    const u16* qrow = Q + (size_t)(q0 + lm) * EMB + h * HD + lh * 8;
#pragma unroll
    for (int dk = 0; dk < 4; ++dk) qf[dk] = load_frag(qrow + dk * 32);
  }

  const u16* krow = Kc + (size_t)lm * EMB + h * HD + lh * 8;
  const u16* vrow = Vt + ((size_t)h * HD + lm) * SEQ + lh * 8;

  v8f ctx[8];
#pragma unroll
  for (int dt = 0; dt < 8; ++dt) ctx[dt] = zero_v8f();
  float m_run = -1e30f, l_run = 0.0f;
  const float SC = 0.08838834764831845f;  // 1/sqrt(128)

  for (int kb = 0; kb < SEQ; kb += 32) {
    // S^T tiles: keys kb..kb+15 (st0) and kb+16..kb+31 (st1), 16 queries.
    v8f st0 = zero_v8f(), st1 = zero_v8f();
#pragma unroll
    for (int dk = 0; dk < 4; ++dk) {
      v16bf a0 = load_frag(krow + (size_t)kb * EMB + dk * 32);
      v16bf a1 = load_frag(krow + (size_t)(kb + 16) * EMB + dk * 32);
      st0 = wmma_bf16(a0, qf[dk], st0);
      st1 = wmma_bf16(a1, qf[dk], st1);
    }

    // Online softmax per query (lane = query; keys split across 8 regs x
    // two tiles, halves merged with one shfl_xor(16)).
    float s0[8], s1[8], tmax = -1e30f;
#pragma unroll
    for (int r = 0; r < 8; ++r) {
      s0[r] = st0[r] * SC;
      s1[r] = st1[r] * SC;
      tmax = fmaxf(tmax, fmaxf(s0[r], s1[r]));
    }
    tmax = fmaxf(tmax, __shfl_xor(tmax, 16, 32));
    float newm = fmaxf(m_run, tmax);
    float corr = __expf(m_run - newm);

    union { v16bf v; u16 s[16]; } pf;
    float psum = 0.0f;
#pragma unroll
    for (int r = 0; r < 8; ++r) {
      float p0 = __expf(s0[r] - newm);
      float p1 = __expf(s1[r] - newm);
      psum += p0 + p1;
      pf.s[r]     = f2bf(p0);  // B elem e<8  <-> key = 8*lh + e
      pf.s[r + 8] = f2bf(p1);  // B elem e>=8 <-> key = 16 + 8*lh + (e-8)
    }
    psum += __shfl_xor(psum, 16, 32);
    l_run = l_run * corr + psum;
    m_run = newm;

#pragma unroll
    for (int dt = 0; dt < 8; ++dt)
#pragma unroll
      for (int r = 0; r < 8; ++r) ctx[dt][r] *= corr;

    // ctx^T[d, q] += Vt_tile(16d x 32keys) x P(32keys x 16q)
#pragma unroll
    for (int dt = 0; dt < 8; ++dt) {
      v16bf av = load_frag(vrow + (size_t)(dt * 16) * SEQ + kb);
      ctx[dt] = wmma_bf16(av, pf.v, ctx[dt]);
    }
  }

  // Normalize and store: lane lm = query, reg r -> d = dt*16 + 8*lh + r
  // (contiguous per dt -> one 16B store each).
  float invl = 1.0f / l_run;
  u16* crow = Ctx + (size_t)(q0 + lm) * EMB + h * HD + lh * 8;
#pragma unroll
  for (int dt = 0; dt < 8; ++dt) {
    union { v4u u; u16 s[8]; } o;
#pragma unroll
    for (int r = 0; r < 8; ++r) o.s[r] = f2bf(ctx[dt][r] * invl);
    *(v4u*)(crow + dt * 16) = o.u;
  }
}

// ---------------------------------------------------------------------------
// Launch
// ---------------------------------------------------------------------------
extern "C" void kernel_launch(void* const* d_in, const int* in_sizes, int n_in,
                              void* d_out, int out_size, void* d_ws,
                              size_t ws_size, hipStream_t stream) {
  (void)in_sizes; (void)n_in; (void)out_size; (void)ws_size;
  const float* X  = (const float*)d_in[0];
  const float* Wq = (const float*)d_in[1];
  const float* Wk = (const float*)d_in[2];
  const float* Wv = (const float*)d_in[3];
  const float* Wo = (const float*)d_in[4];

  char* ws = (char*)d_ws;
  const size_t NB = (size_t)SEQ * EMB * sizeof(u16);  // 8 MB per tensor
  u16* Xb  = (u16*)(ws + 0 * NB);
  u16* Wqb = (u16*)(ws + 1 * NB);
  u16* Wkb = (u16*)(ws + 2 * NB);
  u16* Wvb = (u16*)(ws + 3 * NB);
  u16* Wob = (u16*)(ws + 4 * NB);
  u16* Qb  = (u16*)(ws + 5 * NB);
  u16* Kb  = (u16*)(ws + 6 * NB);
  u16* Vb  = (u16*)(ws + 7 * NB);
  u16* Vt  = (u16*)(ws + 8 * NB);
  u16* Cx  = (u16*)(ws + 9 * NB);

  const int n = SEQ * EMB;           // 4M elements per tensor
  dim3 cgrid(n / 4 / 256);
  cvt_bf16<<<cgrid, 256, 0, stream>>>(X,  Xb,  n);
  cvt_bf16<<<cgrid, 256, 0, stream>>>(Wq, Wqb, n);
  cvt_bf16<<<cgrid, 256, 0, stream>>>(Wk, Wkb, n);
  cvt_bf16<<<cgrid, 256, 0, stream>>>(Wv, Wvb, n);
  cvt_bf16<<<cgrid, 256, 0, stream>>>(Wo, Wob, n);

  dim3 ggrid(EMB / 128, SEQ / 128);
  gemm_nt<true><<<ggrid, 256, 0, stream>>>(Xb, Wqb, Qb, SEQ, EMB, EMB);
  gemm_nt<true><<<ggrid, 256, 0, stream>>>(Xb, Wkb, Kb, SEQ, EMB, EMB);
  gemm_nt<true><<<ggrid, 256, 0, stream>>>(Xb, Wvb, Vb, SEQ, EMB, EMB);

  rope_kernel<<<(SEQ * NH * (HD / 2)) / 256, 256, 0, stream>>>(Qb, Kb);

  transpose_v<<<dim3(SEQ / 32, NH * 4), dim3(32, 8), 0, stream>>>(Vb, Vt);

  attn_kernel<<<dim3(SEQ / 64, NH), 128, 0, stream>>>(Qb, Kb, Vt, Cx);

  gemm_nt<false><<<ggrid, 256, 0, stream>>>(Cx, Wob, d_out, SEQ, EMB, EMB);
}
